// GraphSAGERecommender_85933705659149
// MI455X (gfx1250) — matrix operand order repaired and checked by reading, not compiled, not run on hardware
//
#include <hip/hip_runtime.h>
#include <hip/hip_bf16.h>

typedef __attribute__((ext_vector_type(16))) _Float16 v16h;
typedef __attribute__((ext_vector_type(8)))  float    v8f;

static constexpr int NN  = 100000;   // total nodes
static constexpr int NUU = 60000;    // users (divisible by 16)
static constexpr int HH  = 128;      // hidden == feature dim
static constexpr int EE  = 2000000;  // edges
static constexpr unsigned NEG_INF_ENC = 0x007FFFFFu;    // encode(-inf)
static constexpr int FRAG_HALVES = 512;                 // 32 lanes * 16 halves
static constexpr int MAT_HALVES  = 32 * FRAG_HALVES;    // 32 fragments = 32KB/matrix

// ---- monotonic f32 <-> u32 order-preserving encoding (for atomic max) ----
__device__ __forceinline__ unsigned enc_f32(float f) {
  unsigned u = __float_as_uint(f);
  return (u & 0x80000000u) ? ~u : (u | 0x80000000u);
}
__device__ __forceinline__ float dec_u32(unsigned e) {
  unsigned u = (e & 0x80000000u) ? (e & 0x7FFFFFFFu) : ~e;
  if (u == 0xFF800000u) return 0.0f;   // empty segment (-inf) -> 0, per reference
  return __uint_as_float(u);
}

// ---- A fragment (16x32 f16, MxK): lanes 0-15 hold M=lane, K {0..7}U{16..23};
// lanes 16-31 hold M=lane-16, K {8..15}U{24..31}. ----
__device__ __forceinline__ v16h load_a_f32(const float* __restrict__ tile, int ld,
                                           int kc, int lane) {
  const int m   = lane & 15;
  const int khi = (lane & 16) ? 8 : 0;
  const float* r0 = tile + (size_t)m * ld + kc + khi;
  v16h a;
#pragma unroll
  for (int i = 0; i < 8; ++i) a[i] = (_Float16)r0[i];
#pragma unroll
  for (int i = 0; i < 8; ++i) a[8 + i] = (_Float16)r0[16 + i];
  return a;
}
__device__ __forceinline__ v16h load_a_agg(const unsigned* __restrict__ tile, int ld,
                                           int kc, int lane) {
  const int m   = lane & 15;
  const int khi = (lane & 16) ? 8 : 0;
  const unsigned* r0 = tile + (size_t)m * ld + kc + khi;
  v16h a;
#pragma unroll
  for (int i = 0; i < 8; ++i) a[i] = (_Float16)dec_u32(r0[i]);
#pragma unroll
  for (int i = 0; i < 8; ++i) a[8 + i] = (_Float16)dec_u32(r0[16 + i]);
  return a;
}
// ---- B fragment (32x16 f16, KxN) from f32 W (row-major [HH,HH]), B[k][n] =
// W[(n0+n)*HH + kc+k]. Lanes 0-15: N=lane, K=kc+{0..7}U{16..23}; lanes 16-31: K=kc+{8..15}U{24..31}. ----
__device__ __forceinline__ v16h load_b_wt(const float* __restrict__ w, int n0,
                                          int kc, int lane) {
  const int n  = lane & 15;
  const int kb = kc + ((lane & 16) ? 8 : 0);
  const float* p = w + (size_t)(n0 + n) * HH + kb;
  v16h b;
#pragma unroll
  for (int i = 0; i < 16; ++i) b[i] = (_Float16)p[(i & 7) + ((i >> 3) << 4)];
  return b;
}

// ---- kernel 0: pre-pack 6 weight matrices into f16 B-fragment order ----
// fragment f = kci*8 + nt ; storage: packed[wave*512 + lane*16 .. +15]
__global__ void pack_w_kernel(const float* __restrict__ w0, const float* __restrict__ w1,
                              const float* __restrict__ w2, const float* __restrict__ w3,
                              const float* __restrict__ w4, const float* __restrict__ w5,
                              _Float16* __restrict__ packed) {
  const int wave = blockIdx.x * (blockDim.x >> 5) + (threadIdx.x >> 5);
  const int lane = threadIdx.x & 31;
  if (wave >= 6 * 32) return;
  const int mat = wave >> 5;
  const int f   = wave & 31;
  const int kc  = (f >> 3) * 32;
  const int n0  = (f & 7) * 16;
  const float* w = (mat == 0) ? w0 : (mat == 1) ? w1 : (mat == 2) ? w2
                 : (mat == 3) ? w3 : (mat == 4) ? w4 : w5;
  v16h b = load_b_wt(w, n0, kc, lane);
  *(v16h*)(packed + (size_t)wave * FRAG_HALVES + lane * 16) = b;
}

// ---- kernel 1: typed input projection h0 = x @ W_type^T + b_type ----
__global__ void proj_kernel(const float* __restrict__ x,
                            const _Float16* __restrict__ pw_user,
                            const _Float16* __restrict__ pw_item,
                            const float* __restrict__ b_user,
                            const float* __restrict__ b_item,
                            float* __restrict__ h) {
  __shared__ _Float16 sw[2 * MAT_HALVES];                // 64 KB
  const int tid = threadIdx.x;
  {  // cooperative stage: both matrices (boundary block needs both)
    const float4* g0 = (const float4*)pw_user;
    const float4* g1 = (const float4*)pw_item;
    float4* s0 = (float4*)sw;
    float4* s1 = (float4*)(sw + MAT_HALVES);
#pragma unroll
    for (int i = 0; i < 8; ++i) s0[tid + i * 256] = g0[tid + i * 256];
#pragma unroll
    for (int i = 0; i < 8; ++i) s1[tid + i * 256] = g1[tid + i * 256];
  }
  __syncthreads();

  const int wave = blockIdx.x * (blockDim.x >> 5) + (threadIdx.x >> 5);
  const int lane = threadIdx.x & 31;
  const int strips = NN / 16;
  if (wave >= strips) return;          // wave-uniform: EXEC all-1s for WMMA
  const int row0 = wave * 16;
  const v16h*  wb   = (const v16h*)((row0 < NUU) ? sw : sw + MAT_HALVES);
  const float* bias = (row0 < NUU) ? b_user : b_item;

  v8f c[8] = {};
  for (int kci = 0; kci < 4; ++kci) {
    v16h a = load_a_f32(x + (size_t)row0 * HH, HH, kci * 32, lane);
#pragma unroll
    for (int nt = 0; nt < 8; ++nt) {
      v16h b = wb[(kci * 8 + nt) * 32 + lane];           // ds_load, no cvt
      c[nt] = __builtin_amdgcn_wmma_f32_16x16x32_f16(false, a, false, b,
                                                     (short)0, c[nt], false, false);
    }
  }
  const int col   = lane & 15;
  const int rbase = (lane & 16) ? 8 : 0;
#pragma unroll
  for (int r = 0; r < 8; ++r) {
    const size_t row = (size_t)(row0 + rbase + r);
#pragma unroll
    for (int nt = 0; nt < 8; ++nt) {
      const int n = nt * 16 + col;
      h[row * HH + n] = c[nt][r] + bias[n];
    }
  }
}

// ---- kernel 2: reset aggregation buffer to encode(-inf) ----
__global__ void init_agg_kernel(unsigned* __restrict__ agg) {
  const int i = blockIdx.x * blockDim.x + threadIdx.x;
  if (i < NN * HH) agg[i] = NEG_INF_ENC;
}

// ---- kernel 3: per-edge gather + atomic max scatter (L2-resident) ----
__global__ void scatter_kernel(const float* __restrict__ h, const int* __restrict__ src,
                               const int* __restrict__ dst, unsigned* __restrict__ agg) {
  const long long t = (long long)blockIdx.x * blockDim.x + threadIdx.x;
  const int e = (int)(t >> 5);
  if (e >= EE) return;
  const int c4 = ((int)t & 31) * 4;    // 4 channels/lane, 128 per wave/edge
  const int s = src[e];
  const int d = dst[e];
  const float4 v = *(const float4*)(h + (size_t)s * HH + c4);
  unsigned* ap = agg + (size_t)d * HH + c4;
  atomicMax(ap + 0, enc_f32(v.x));
  atomicMax(ap + 1, enc_f32(v.y));
  atomicMax(ap + 2, enc_f32(v.z));
  atomicMax(ap + 3, enc_f32(v.w));
}

// ---- kernel 4: out = act( h + dec(agg) @ w_l^T + b_l + h @ w_r^T ) ----
__global__ void sage_kernel(const float* __restrict__ h, const unsigned* __restrict__ agg,
                            const _Float16* __restrict__ pw_l,
                            const _Float16* __restrict__ pw_r,
                            const float* __restrict__ b_l,
                            float* __restrict__ out, int do_relu) {
  __shared__ _Float16 sw[2 * MAT_HALVES];                // 64 KB: w_l | w_r
  const int tid = threadIdx.x;
  {
    const float4* g0 = (const float4*)pw_l;
    const float4* g1 = (const float4*)pw_r;
    float4* s0 = (float4*)sw;
    float4* s1 = (float4*)(sw + MAT_HALVES);
#pragma unroll
    for (int i = 0; i < 8; ++i) s0[tid + i * 256] = g0[tid + i * 256];
#pragma unroll
    for (int i = 0; i < 8; ++i) s1[tid + i * 256] = g1[tid + i * 256];
  }
  __syncthreads();

  const int wave = blockIdx.x * (blockDim.x >> 5) + (threadIdx.x >> 5);
  const int lane = threadIdx.x & 31;
  const int strips = NN / 16;
  if (wave >= strips) return;
  const int row0 = wave * 16;
  const v16h* wl = (const v16h*)sw;
  const v16h* wr = (const v16h*)(sw + MAT_HALVES);

  v8f c[8] = {};
  for (int kci = 0; kci < 4; ++kci) {
    v16h a_h = load_a_f32(h   + (size_t)row0 * HH, HH, kci * 32, lane);
    v16h a_g = load_a_agg(agg + (size_t)row0 * HH, HH, kci * 32, lane);
#pragma unroll
    for (int nt = 0; nt < 8; ++nt) {
      v16h bl = wl[(kci * 8 + nt) * 32 + lane];
      v16h br = wr[(kci * 8 + nt) * 32 + lane];
      c[nt] = __builtin_amdgcn_wmma_f32_16x16x32_f16(false, a_g, false, bl,
                                                     (short)0, c[nt], false, false);
      c[nt] = __builtin_amdgcn_wmma_f32_16x16x32_f16(false, a_h, false, br,
                                                     (short)0, c[nt], false, false);
    }
  }
  const int col   = lane & 15;
  const int rbase = (lane & 16) ? 8 : 0;
#pragma unroll
  for (int r = 0; r < 8; ++r) {
    const size_t row = (size_t)(row0 + rbase + r);
#pragma unroll
    for (int nt = 0; nt < 8; ++nt) {
      const int n = nt * 16 + col;
      float v = h[row * HH + n] + c[nt][r] + b_l[n];   // residual + conv + bias
      if (do_relu) v = fmaxf(v, 0.0f);
      out[row * HH + n] = v;                           // in-place safe (own rows only)
    }
  }
}

extern "C" void kernel_launch(void* const* d_in, const int* in_sizes, int n_in,
                              void* d_out, int out_size, void* d_ws, size_t ws_size,
                              hipStream_t stream) {
  (void)in_sizes; (void)n_in; (void)out_size; (void)ws_size;
  const float* x      = (const float*)d_in[0];
  const int*   ei     = (const int*)  d_in[1];
  /* d_in[2] node_type: unused (nodes sorted, NU divisible by 16) */
  const float* w_user = (const float*)d_in[3];
  const float* b_user = (const float*)d_in[4];
  const float* w_item = (const float*)d_in[5];
  const float* b_item = (const float*)d_in[6];
  const float* w_l0   = (const float*)d_in[7];
  const float* b_l0   = (const float*)d_in[8];
  const float* w_r0   = (const float*)d_in[9];
  const float* w_l1   = (const float*)d_in[10];
  const float* b_l1   = (const float*)d_in[11];
  const float* w_r1   = (const float*)d_in[12];

  float*     out = (float*)d_out;                             // also layer-1 hidden
  float*     h0  = (float*)d_ws;                              // 100000*128 f32
  unsigned*  agg = (unsigned*)((char*)d_ws + (size_t)NN * HH * sizeof(float));
  _Float16*  pw  = (_Float16*)((char*)d_ws + 2 * (size_t)NN * HH * 4);  // 6*32KB
  const int* src = ei;
  const int* dst = ei + EE;

  const dim3 blk(256);
  const int strips     = NN / 16;                  // 6250 GEMM row-strips
  const int gemmBlocks = (strips + 7) / 8;         // 8 waves / 256-thread block
  const int initBlocks = (NN * HH + 255) / 256;
  const int scatBlocks = (int)(((long long)EE * 32 + 255) / 256);

  // pack weights to f16 fragment layout: [user, item, l0, r0, l1, r1]
  pack_w_kernel<<<6, blk, 0, stream>>>(w_user, w_item, w_l0, w_r0, w_l1, w_r1, pw);
  const _Float16* pw_user = pw + 0 * MAT_HALVES;
  const _Float16* pw_item = pw + 1 * MAT_HALVES;
  const _Float16* pw_l0   = pw + 2 * MAT_HALVES;
  const _Float16* pw_r0   = pw + 3 * MAT_HALVES;
  const _Float16* pw_l1   = pw + 4 * MAT_HALVES;
  const _Float16* pw_r1   = pw + 5 * MAT_HALVES;

  // input projection -> h0
  proj_kernel<<<gemmBlocks, blk, 0, stream>>>(x, pw_user, pw_item, b_user, b_item, h0);

  // layer 0: max-aggregate + GEMMs + residual + ReLU -> d_out
  init_agg_kernel<<<initBlocks, blk, 0, stream>>>(agg);
  scatter_kernel<<<scatBlocks, blk, 0, stream>>>(h0, src, dst, agg);
  sage_kernel<<<gemmBlocks, blk, 0, stream>>>(h0, agg, pw_l0, pw_r0, b_l0, out, 1);

  // layer 1: same on d_out, no ReLU, in-place final result
  init_agg_kernel<<<initBlocks, blk, 0, stream>>>(agg);
  scatter_kernel<<<scatBlocks, blk, 0, stream>>>(out, src, dst, agg);
  sage_kernel<<<gemmBlocks, blk, 0, stream>>>(out, agg, pw_l1, pw_r1, b_l1, out, 0);
}
